// MSAB2_16381005267663
// MI455X (gfx1250) — matrix operand order repaired and checked by reading, not compiled, not run on hardware
//
#include <hip/hip_runtime.h>
#include <hip/hip_bf16.h>
#include <math.h>

typedef __attribute__((ext_vector_type(16))) _Float16 v16h;
typedef __attribute__((ext_vector_type(8)))  _Float16 v8h;
typedef __attribute__((ext_vector_type(8)))  float    v8f;

#define DEV __device__ __forceinline__

// gfx1250 async global->LDS path (ASYNCcnt), guarded so the file always compiles
#if defined(__has_builtin)
#  if __has_builtin(__builtin_amdgcn_global_load_async_to_lds_b128) && \
      __has_builtin(__builtin_amdgcn_s_wait_asynccnt)
#    define USE_ASYNC_LDS 1
#  endif
#endif
#ifndef USE_ASYNC_LDS
#  define USE_ASYNC_LDS 0
#endif

#if USE_ASYNC_LDS
typedef int v4i_ __attribute__((vector_size(16)));
typedef __attribute__((address_space(1))) v4i_* gas_ptr;   // int4 __device__ *
typedef __attribute__((address_space(3))) v4i_* las_ptr;   // int4 __shared__ *
#endif

DEV float geluf(float v) { return 0.5f * v * (1.0f + erff(v * 0.70710678118654752440f)); }

DEV v8f wmma16(v16h a, v16h b, v8f c) {
    // D = A(16x32 f16) x B(32x16 f16) + C(16x16 f32)
    return __builtin_amdgcn_wmma_f32_16x16x32_f16(false, a, false, b, (short)0, c, false, false);
}

// ---------------------------------------------------------------------------
// Generic WMMA GEMM: out[M,N] = act(A[M,K] @ Bw[K,N] + bias (+ res))
// block = 256 threads (8 waves); wave computes 16 rows x 64 cols; grid (M/128, N/64)
// B tile staged to LDS with double-buffered async global->LDS loads (ASYNCcnt)
// ---------------------------------------------------------------------------
template <int OUT_F16, int ACT>
__global__ __launch_bounds__(256) void gemm_f16_k(
    const _Float16* __restrict__ A, const _Float16* __restrict__ Bw,
    const float* __restrict__ bias, const float* __restrict__ res,
    void* __restrict__ outp, int N, int K)
{
    __shared__ _Float16 Bs[2][32][72];        // double-buffered 32(K) x 64 cols, padded
    const int lane = threadIdx.x & 31;
    const int wave = threadIdx.x >> 5;
    const int hs   = lane >> 4;               // half-wave select
    const int ln   = lane & 15;
    const size_t row0 = (size_t)blockIdx.x * 128 + wave * 16;
    const int col0 = blockIdx.y * 64;
    const int br = threadIdx.x >> 3;          // B tile row this thread copies
    const int bc = (threadIdx.x & 7) * 8;     // B tile col start (8 halves = 16B)

    v8f acc[4] = {};

#if USE_ASYNC_LDS
    // prologue: async-stage first tile
    __builtin_amdgcn_global_load_async_to_lds_b128(
        (gas_ptr)(Bw + (size_t)br * N + col0 + bc),
        (las_ptr)(&Bs[0][br][bc]), 0, 0);
    __builtin_amdgcn_s_wait_asynccnt(0);
    __syncthreads();
    int buf = 0;
    for (int k0 = 0; k0 < K; k0 += 32) {
        if (k0 + 32 < K)  // prefetch next tile into the other buffer (async)
            __builtin_amdgcn_global_load_async_to_lds_b128(
                (gas_ptr)(Bw + (size_t)(k0 + 32 + br) * N + col0 + bc),
                (las_ptr)(&Bs[buf ^ 1][br][bc]), 0, 0);
        v16h a;
        const _Float16* ar = A + (row0 + ln) * (size_t)K + k0 + hs * 8;
        #pragma unroll
        for (int i = 0; i < 8; ++i) a[i] = ar[i];
        #pragma unroll
        for (int i = 0; i < 8; ++i) a[8 + i] = ar[16 + i];
        #pragma unroll
        for (int nt = 0; nt < 4; ++nt) {
            v16h bf;
            #pragma unroll
            for (int i = 0; i < 16; ++i) bf[i] = Bs[buf][hs * 8 + (i & 7) + ((i >> 3) << 4)][nt * 16 + ln];
            acc[nt] = wmma16(a, bf, acc[nt]);
        }
        if (k0 + 32 < K) __builtin_amdgcn_s_wait_asynccnt(0);
        __syncthreads();
        buf ^= 1;
    }
#else
    for (int k0 = 0; k0 < K; k0 += 32) {
        *(v8h*)(&Bs[0][br][bc]) = *(const v8h*)(Bw + (size_t)(k0 + br) * N + col0 + bc);
        if (k0 + 32 < K)
            __builtin_prefetch(Bw + (size_t)(k0 + 32 + br) * N + col0 + bc, 0, 1);
        __syncthreads();
        v16h a;
        const _Float16* ar = A + (row0 + ln) * (size_t)K + k0 + hs * 8;
        #pragma unroll
        for (int i = 0; i < 8; ++i) a[i] = ar[i];
        #pragma unroll
        for (int i = 0; i < 8; ++i) a[8 + i] = ar[16 + i];
        #pragma unroll
        for (int nt = 0; nt < 4; ++nt) {
            v16h bf;
            #pragma unroll
            for (int i = 0; i < 16; ++i) bf[i] = Bs[0][hs * 8 + (i & 7) + ((i >> 3) << 4)][nt * 16 + ln];
            acc[nt] = wmma16(a, bf, acc[nt]);
        }
        __syncthreads();
    }
#endif

    #pragma unroll
    for (int nt = 0; nt < 4; ++nt) {
        int n = col0 + nt * 16 + ln;
        float bv = bias ? bias[n] : 0.0f;
        #pragma unroll
        for (int r = 0; r < 8; ++r) {
            size_t m = row0 + r + hs * 8;
            float v = acc[nt][r] + bv;
            if (res) v += res[m * (size_t)N + n];
            if (ACT) v = geluf(v);
            if (OUT_F16) ((_Float16*)outp)[m * (size_t)N + n] = (_Float16)v;
            else         ((float*)outp)[m * (size_t)N + n] = v;
        }
    }
}

// ---------------------------------------------------------------------------
// LN1 + roll(-4,-4) + 8x8 window partition -> f16 [B*1024 windows][64][128]
// ---------------------------------------------------------------------------
__global__ __launch_bounds__(256) void ln1_window_k(
    const float* __restrict__ x, const float* __restrict__ g,
    const float* __restrict__ bt, _Float16* __restrict__ xw)
{
    int bw = blockIdx.x;                 // b*1024 + window
    int b = bw >> 10, win = bw & 1023;
    int hy = win >> 5, wx = win & 31;
    int wave = threadIdx.x >> 5, lane = threadIdx.x & 31;
    for (int pp = wave; pp < 64; pp += 8) {
        int i = pp >> 3, j = pp & 7;
        int y  = (hy * 8 + i + 4) & 255;
        int xx = (wx * 8 + j + 4) & 255;
        const float* px = x + (((size_t)b * 256 + y) * 256 + xx) * 128;
        float v[4]; float s = 0.f;
        #pragma unroll
        for (int t = 0; t < 4; ++t) { v[t] = px[lane + 32 * t]; s += v[t]; }
        #pragma unroll
        for (int o = 1; o < 32; o <<= 1) s += __shfl_xor(s, o, 32);
        float mu = s * (1.0f / 128.0f);
        float ss = 0.f;
        #pragma unroll
        for (int t = 0; t < 4; ++t) { float d = v[t] - mu; ss += d * d; }
        #pragma unroll
        for (int o = 1; o < 32; o <<= 1) ss += __shfl_xor(ss, o, 32);
        float inv = rsqrtf(ss * (1.0f / 128.0f) + 1e-5f);
        _Float16* op = xw + ((size_t)bw * 64 + pp) * 128;
        #pragma unroll
        for (int t = 0; t < 4; ++t) {
            int c = lane + 32 * t;
            op[c] = (_Float16)((v[t] - mu) * inv * g[c] + bt[c]);
        }
    }
}

// Generic LN -> f16 (pixel order), one pixel per wave
__global__ __launch_bounds__(256) void ln_f16_k(
    const float* __restrict__ x, const float* __restrict__ g,
    const float* __restrict__ bt, _Float16* __restrict__ out)
{
    int pix = blockIdx.x * 8 + (threadIdx.x >> 5);
    int lane = threadIdx.x & 31;
    const float* px = x + (size_t)pix * 128;
    float v[4]; float s = 0.f;
    #pragma unroll
    for (int t = 0; t < 4; ++t) { v[t] = px[lane + 32 * t]; s += v[t]; }
    #pragma unroll
    for (int o = 1; o < 32; o <<= 1) s += __shfl_xor(s, o, 32);
    float mu = s * (1.0f / 128.0f);
    float ss = 0.f;
    #pragma unroll
    for (int t = 0; t < 4; ++t) { float d = v[t] - mu; ss += d * d; }
    #pragma unroll
    for (int o = 1; o < 32; o <<= 1) ss += __shfl_xor(ss, o, 32);
    float inv = rsqrtf(ss * (1.0f / 128.0f) + 1e-5f);
    _Float16* op = out + (size_t)pix * 128;
    #pragma unroll
    for (int t = 0; t < 4; ++t) {
        int c = lane + 32 * t;
        op[c] = (_Float16)((v[t] - mu) * inv * g[c] + bt[c]);
    }
}

// ---------------------------------------------------------------------------
// Window attention: one block per window (128 thr = 4 waves, wave = head)
// qkv rows: [b*1024+win]*64+p, cols: which*128 + head*32 + d
// ---------------------------------------------------------------------------
__global__ __launch_bounds__(128) void attn_win_k(
    const _Float16* __restrict__ qkv, const float* __restrict__ relf,
    _Float16* __restrict__ owin)
{
    int bw = blockIdx.x;
    int win = bw & 1023;
    int hw = win >> 5, ww = win & 31;
    int h = threadIdx.x >> 5;
    int lane = threadIdx.x & 31;
    int ln = lane & 15, hs = lane >> 4;
    const size_t m0 = (size_t)bw * 64;
    const _Float16* qb = qkv + m0 * 384 + h * 32;
    const _Float16* kb = qkv + m0 * 384 + 128 + h * 32;
    const _Float16* vb = qkv + m0 * 384 + 256 + h * 32;

    // sim = q @ k^T   (64x64 = 4x4 WMMA tiles, K=32)
    v16h afrag[4];
    #pragma unroll
    for (int mt = 0; mt < 4; ++mt) {
        const _Float16* ar = qb + (size_t)(mt * 16 + ln) * 384 + hs * 8;
        #pragma unroll
        for (int i = 0; i < 8; ++i) afrag[mt][i] = ar[i];
        #pragma unroll
        for (int i = 0; i < 8; ++i) afrag[mt][8 + i] = ar[16 + i];
    }
    v8f acc[4][4] = {};
    #pragma unroll
    for (int nt = 0; nt < 4; ++nt) {
        v16h bf;  // B[d][token]: b[i] = k[token=nt*16+ln][d=hs*8+(i&7)+(i>>3)*16]
        const _Float16* br = kb + (size_t)(nt * 16 + ln) * 384 + hs * 8;
        #pragma unroll
        for (int i = 0; i < 16; ++i) bf[i] = br[(i & 7) + ((i >> 3) << 4)];
        #pragma unroll
        for (int mt = 0; mt < 4; ++mt) acc[mt][nt] = wmma16(afrag[mt], bf, acc[mt][nt]);
    }

    // scale + rel bias + shift mask + row softmax (rows split per half-wave)
    const float scale = 0.17677669529663687f;  // 1/sqrt(32)
    bool lastH = (hw == 31), lastW = (ww == 31);
    const float* relh = relf + (h << 12);
    #pragma unroll
    for (int mt = 0; mt < 4; ++mt) {
        #pragma unroll
        for (int nt = 0; nt < 4; ++nt) {
            int n = nt * 16 + ln;
            int i2 = n >> 3, j2 = n & 7;
            #pragma unroll
            for (int r = 0; r < 8; ++r) {
                int p = mt * 16 + r + hs * 8;
                int i1 = p >> 3, j1 = p & 7;
                float v = acc[mt][nt][r] * scale + relh[p * 64 + n];
                bool msk = (lastH && ((i1 < 4) != (i2 < 4))) ||
                           (lastW && ((j1 < 4) != (j2 < 4)));
                acc[mt][nt][r] = msk ? -1e30f : v;
            }
        }
        #pragma unroll
        for (int r = 0; r < 8; ++r) {
            float mx = -1e30f;
            #pragma unroll
            for (int nt = 0; nt < 4; ++nt) mx = fmaxf(mx, acc[mt][nt][r]);
            #pragma unroll
            for (int o = 1; o < 16; o <<= 1) mx = fmaxf(mx, __shfl_xor(mx, o, 32));
            float s = 0.f;
            #pragma unroll
            for (int nt = 0; nt < 4; ++nt) {
                float e = __expf(acc[mt][nt][r] - mx);
                acc[mt][nt][r] = e; s += e;
            }
            #pragma unroll
            for (int o = 1; o < 16; o <<= 1) s += __shfl_xor(s, o, 32);
            float inv = 1.0f / s;
            #pragma unroll
            for (int nt = 0; nt < 4; ++nt) acc[mt][nt][r] *= inv;
        }
    }

    // probs -> LDS (f16), then o = probs @ v (64x32, K=64)
    __shared__ _Float16 probs[4][64][72];
    #pragma unroll
    for (int mt = 0; mt < 4; ++mt)
        #pragma unroll
        for (int nt = 0; nt < 4; ++nt)
            #pragma unroll
            for (int r = 0; r < 8; ++r)
                probs[h][mt * 16 + r + hs * 8][nt * 16 + ln] = (_Float16)acc[mt][nt][r];
    __syncthreads();

    v8f oacc[4][2] = {};
    #pragma unroll
    for (int kt = 0; kt < 2; ++kt) {
        v16h pa[4];
        #pragma unroll
        for (int mt = 0; mt < 4; ++mt) {
            const _Float16* pr = &probs[h][mt * 16 + ln][kt * 32 + hs * 8];
            #pragma unroll
            for (int i = 0; i < 8; ++i) pa[mt][i] = pr[i];
            #pragma unroll
            for (int i = 0; i < 8; ++i) pa[mt][8 + i] = pr[16 + i];
        }
        #pragma unroll
        for (int nt = 0; nt < 2; ++nt) {
            v16h bf;  // B[token][d]: b[i] = v[kt*32+hs*8+(i&7)+(i>>3)*16][nt*16+ln]
            const _Float16* br = vb + (size_t)(kt * 32 + hs * 8) * 384 + nt * 16 + ln;
            #pragma unroll
            for (int i = 0; i < 16; ++i) bf[i] = br[(size_t)((i & 7) + ((i >> 3) << 4)) * 384];
            #pragma unroll
            for (int mt = 0; mt < 4; ++mt) oacc[mt][nt] = wmma16(pa[mt], bf, oacc[mt][nt]);
        }
    }
    #pragma unroll
    for (int mt = 0; mt < 4; ++mt)
        #pragma unroll
        for (int nt = 0; nt < 2; ++nt)
            #pragma unroll
            for (int r = 0; r < 8; ++r) {
                int p = mt * 16 + r + hs * 8;
                int d = nt * 16 + ln;
                owin[(m0 + p) * 128 + h * 32 + d] = (_Float16)oacc[mt][nt][r];
            }
}

// ---------------------------------------------------------------------------
// Channel attention helpers
// ---------------------------------------------------------------------------
__global__ __launch_bounds__(128) void sumsq_k(
    const _Float16* __restrict__ qi, const _Float16* __restrict__ ki,
    float* __restrict__ qss, float* __restrict__ kss)
{
    int blk = blockIdx.x; int b = blk >> 7; int chunk = blk & 127;
    int c = threadIdx.x;
    size_t base = ((size_t)b * 65536 + (size_t)chunk * 512) * 128 + c;
    float sq = 0.f, sk = 0.f;
    for (int p = 0; p < 512; ++p) {
        float q = (float)qi[base + (size_t)p * 128];
        float k = (float)ki[base + (size_t)p * 128];
        sq += q * q; sk += k * k;
    }
    atomicAdd(&qss[b * 128 + c], sq);
    atomicAdd(&kss[b * 128 + c], sk);
}

__global__ __launch_bounds__(256) void gram_k(
    const _Float16* __restrict__ qi, const _Float16* __restrict__ ki,
    float* __restrict__ G)
{
    int blk = blockIdx.x;                       // (b,h,chunk of 1024 pixels)
    int chunk = blk & 63; int h = (blk >> 6) & 3; int b = blk >> 8;
    __shared__ float qs[64][33], ks[64][33];
    int tid = threadIdx.x;
    int pair0 = tid * 4;
    float accv[4] = {0.f, 0.f, 0.f, 0.f};
    size_t nbase = (size_t)b * 65536 + (size_t)chunk * 1024;
    for (int it = 0; it < 16; ++it) {
        #pragma unroll
        for (int l = 0; l < 8; ++l) {
            int e = tid * 8 + l;
            int p = e >> 5, cc = e & 31;
            size_t idx = (nbase + it * 64 + p) * 128 + h * 32 + cc;
            qs[p][cc] = (float)qi[idx];
            ks[p][cc] = (float)ki[idx];
        }
        __syncthreads();
        #pragma unroll
        for (int u = 0; u < 4; ++u) {
            int de = pair0 + u; int d = de >> 5, e = de & 31;
            float s = accv[u];
            for (int p = 0; p < 64; ++p) s += ks[p][d] * qs[p][e];
            accv[u] = s;
        }
        __syncthreads();
    }
    #pragma unroll
    for (int u = 0; u < 4; ++u)
        atomicAdd(&G[((size_t)(b * 4 + h) << 10) + pair0 + u], accv[u]);
}

__global__ void attn_softmax_k(
    const float* __restrict__ G, const float* __restrict__ qss,
    const float* __restrict__ kss, const float* __restrict__ rescale,
    float* __restrict__ attn)
{
    int bh = blockIdx.x; int b = bh >> 2, h = bh & 3;
    int d = threadIdx.x;                       // 32 threads
    float kn = fmaxf(sqrtf(kss[b * 128 + h * 32 + d]), 1e-12f);
    float rs = rescale[h];
    float vals[32]; float mx = -1e30f;
    for (int e = 0; e < 32; ++e) {
        float qn = fmaxf(sqrtf(qss[b * 128 + h * 32 + e]), 1e-12f);
        float v = G[(bh << 10) + (d << 5) + e] / (kn * qn) * rs;
        vals[e] = v; mx = fmaxf(mx, v);
    }
    float s = 0.f;
    for (int e = 0; e < 32; ++e) { vals[e] = __expf(vals[e] - mx); s += vals[e]; }
    float inv = 1.0f / s;
    for (int e = 0; e < 32; ++e) attn[(bh << 10) + (d << 5) + e] = vals[e] * inv;
}

__global__ __launch_bounds__(256) void apply_attn_k(
    const _Float16* __restrict__ vi, const float* __restrict__ attn,
    _Float16* __restrict__ xo)
{
    __shared__ _Float16 vrow[2][128];
    int half = threadIdx.x >> 7;
    int c = threadIdx.x & 127;
    int pix = blockIdx.x * 2 + half;
    int b = pix >> 16;
    vrow[half][c] = vi[(size_t)pix * 128 + c];
    __syncthreads();
    int h = c >> 5, d = c & 31;
    const float* arow = attn + (((size_t)(b * 4 + h) << 5) + d) * 32;
    float s = 0.f;
    #pragma unroll
    for (int e = 0; e < 32; ++e) s += arow[e] * (float)vrow[half][h * 32 + e];
    xo[(size_t)pix * 128 + c] = (_Float16)s;
}

// ---------------------------------------------------------------------------
// Depthwise 3x3 conv (SAME). GELU_OUT=1: f16 out with gelu; else spec += conv
// ---------------------------------------------------------------------------
template <int GELU_OUT>
__global__ __launch_bounds__(256) void dwconv_k(
    const _Float16* __restrict__ in, const float* __restrict__ wk,
    _Float16* __restrict__ outh, float* __restrict__ addf)
{
    int pix = blockIdx.x * 2 + (threadIdx.x >> 7);
    int c = threadIdx.x & 127;
    int b = pix >> 16; int rem = pix & 65535; int y = rem >> 8; int x = rem & 255;
    float s = 0.f;
    #pragma unroll
    for (int ky = 0; ky < 3; ++ky) {
        int yy = y + ky - 1; if (yy < 0 || yy > 255) continue;
        #pragma unroll
        for (int kx = 0; kx < 3; ++kx) {
            int xx = x + kx - 1; if (xx < 0 || xx > 255) continue;
            s += wk[(ky * 3 + kx) * 128 + c] *
                 (float)in[(((size_t)b * 256 + yy) * 256 + xx) * 128 + c];
        }
    }
    if (GELU_OUT) outh[(size_t)pix * 128 + c] = (_Float16)geluf(s);
    else          addf[(size_t)pix * 128 + c] += s;
}

// x2 = x + spa(un-window+roll) + spec ; h3 = ln3(x2)
__global__ __launch_bounds__(256) void combine_ln3_k(
    const float* __restrict__ x, const float* __restrict__ spaw,
    const float* __restrict__ spec, const float* __restrict__ g,
    const float* __restrict__ bt, float* __restrict__ x2,
    _Float16* __restrict__ h3)
{
    int pix = blockIdx.x * 8 + (threadIdx.x >> 5);
    int lane = threadIdx.x & 31;
    int b = pix >> 16; int rem = pix & 65535; int y = rem >> 8; int xx = rem & 255;
    int yp = (y - 4) & 255, xp = (xx - 4) & 255;
    int win = (yp >> 3) * 32 + (xp >> 3);
    int p = (yp & 7) * 8 + (xp & 7);
    size_t spaBase = (((size_t)b * 1024 + win) * 64 + p) * 128;
    size_t base = (size_t)pix * 128;
    float v[4]; float s = 0.f;
    #pragma unroll
    for (int t = 0; t < 4; ++t) {
        int c = lane + 32 * t;
        float val = x[base + c] + spaw[spaBase + c] + spec[base + c];
        v[t] = val; s += val;
    }
    #pragma unroll
    for (int o = 1; o < 32; o <<= 1) s += __shfl_xor(s, o, 32);
    float mu = s * (1.0f / 128.0f);
    float ss = 0.f;
    #pragma unroll
    for (int t = 0; t < 4; ++t) { float d = v[t] - mu; ss += d * d; }
    #pragma unroll
    for (int o = 1; o < 32; o <<= 1) ss += __shfl_xor(ss, o, 32);
    float inv = rsqrtf(ss * (1.0f / 128.0f) + 1e-5f);
    #pragma unroll
    for (int t = 0; t < 4; ++t) {
        int c = lane + 32 * t;
        x2[base + c] = v[t];
        h3[base + c] = (_Float16)((v[t] - mu) * inv * g[c] + bt[c]);
    }
}

// ---------------------------------------------------------------------------
// Small utility kernels
// ---------------------------------------------------------------------------
__global__ void f32_to_f16_k(const float* __restrict__ in, _Float16* __restrict__ out, int n) {
    int i = blockIdx.x * 256 + threadIdx.x;
    if (i < n) out[i] = (_Float16)in[i];
}
__global__ void zerof_k(float* __restrict__ p, int n) {
    int i = blockIdx.x * 256 + threadIdx.x;
    if (i < n) p[i] = 0.0f;
}
__global__ void relfull_k(const float* __restrict__ rel_pos, float* __restrict__ relf) {
    int idx = blockIdx.x * 256 + threadIdx.x;
    if (idx >= 4 * 64 * 64) return;
    int h = idx >> 12; int p = (idx >> 6) & 63; int q = idx & 63;
    int ri = (p >> 3) - (q >> 3) + 7;
    int rj = (p & 7) - (q & 7) + 7;
    relf[idx] = rel_pos[(h * 15 + ri) * 15 + rj];
}

// ---------------------------------------------------------------------------
extern "C" void kernel_launch(void* const* d_in, const int* in_sizes, int n_in,
                              void* d_out, int out_size, void* d_ws, size_t ws_size,
                              hipStream_t stream)
{
    (void)in_sizes; (void)n_in; (void)out_size; (void)ws_size;
    const float* x      = (const float*)d_in[0];
    const float* ln1_g  = (const float*)d_in[1];
    const float* ln1_b  = (const float*)d_in[2];
    const float* ln2_g  = (const float*)d_in[3];
    const float* ln2_b  = (const float*)d_in[4];
    const float* ln3_g  = (const float*)d_in[5];
    const float* ln3_b  = (const float*)d_in[6];
    const float* w_qkv  = (const float*)d_in[7];
    const float* b_qkv  = (const float*)d_in[8];
    const float* rel_pos= (const float*)d_in[9];
    const float* w_wlin = (const float*)d_in[10];
    const float* b_wlin = (const float*)d_in[11];
    const float* w_q    = (const float*)d_in[12];
    const float* w_k    = (const float*)d_in[13];
    const float* w_v    = (const float*)d_in[14];
    const float* rescale= (const float*)d_in[15];
    const float* w_proj = (const float*)d_in[16];
    const float* b_proj = (const float*)d_in[17];
    const float* conv1_w= (const float*)d_in[18];
    const float* conv2_w= (const float*)d_in[19];
    const float* w_mlp1 = (const float*)d_in[20];
    const float* b_mlp1 = (const float*)d_in[21];
    const float* w_mlp2 = (const float*)d_in[22];
    const float* b_mlp2 = (const float*)d_in[23];

    char* ws = (char*)d_ws;
    _Float16* xw_h  = (_Float16*)(ws + 0ULL);               // 64 MiB (dead after qkv)
    _Float16* qkv_h = (_Float16*)(ws + 67108864ULL);        // 192 MiB (dead after attn)
    _Float16* m1_h  = (_Float16*)(ws + 0ULL);               // 256 MiB reuse
    _Float16* xn_h  = (_Float16*)(ws + 268435456ULL);       // 64 MiB
    _Float16* sbuf_h= (_Float16*)(ws + 335544320ULL);       // 64 MiB: owin/xo/t/h3 (serial reuse)
    float*    spa_w = (float*)   (ws + 402653184ULL);       // 128 MiB
    _Float16* qi_h  = (_Float16*)(ws + 536870912ULL);       // 64 MiB
    _Float16* ki_h  = (_Float16*)(ws + 603979776ULL);       // 64 MiB
    float*    x2f   = (float*)   (ws + 536870912ULL);       // 128 MiB reuse of qi+ki
    _Float16* vi_h  = (_Float16*)(ws + 671088640ULL);       // 64 MiB
    float*    specf = (float*)   (ws + 738197504ULL);       // 128 MiB
    char* sm = ws + 872415232ULL;
    float* G    = (float*)(sm + 0);          // 16384 f
    float* qss  = (float*)(sm + 65536);      // 512 f
    float* kss  = (float*)(sm + 67584);      // 512 f
    float* attn = (float*)(sm + 69632);      // 16384 f
    float* relf = (float*)(sm + 135168);     // 16384 f
    _Float16* wqkv16  = (_Float16*)(sm + 200704);
    _Float16* wwlin16 = (_Float16*)(sm + 299008);
    _Float16* wq16    = (_Float16*)(sm + 331776);
    _Float16* wk16    = (_Float16*)(sm + 364544);
    _Float16* wv16    = (_Float16*)(sm + 397312);
    _Float16* wproj16 = (_Float16*)(sm + 430080);
    _Float16* wmlp1_16= (_Float16*)(sm + 462848);
    _Float16* wmlp2_16= (_Float16*)(sm + 593920);

    // 0) prep: zero accumulators, f16 weights, rel-pos table
    zerof_k<<<(17408 + 255) / 256, 256, 0, stream>>>(G, 17408);
    f32_to_f16_k<<<192, 256, 0, stream>>>(w_qkv,  wqkv16,  49152);
    f32_to_f16_k<<<64,  256, 0, stream>>>(w_wlin, wwlin16, 16384);
    f32_to_f16_k<<<64,  256, 0, stream>>>(w_q,    wq16,    16384);
    f32_to_f16_k<<<64,  256, 0, stream>>>(w_k,    wk16,    16384);
    f32_to_f16_k<<<64,  256, 0, stream>>>(w_v,    wv16,    16384);
    f32_to_f16_k<<<64,  256, 0, stream>>>(w_proj, wproj16, 16384);
    f32_to_f16_k<<<256, 256, 0, stream>>>(w_mlp1, wmlp1_16, 65536);
    f32_to_f16_k<<<256, 256, 0, stream>>>(w_mlp2, wmlp2_16, 65536);
    relfull_k<<<64, 256, 0, stream>>>(rel_pos, relf);

    // 1) window branch
    ln1_window_k<<<4096, 256, 0, stream>>>(x, ln1_g, ln1_b, xw_h);
    { dim3 g(2048, 6);
      gemm_f16_k<1, 0><<<g, 256, 0, stream>>>(xw_h, wqkv16, b_qkv, nullptr, qkv_h, 384, 128); }
    attn_win_k<<<4096, 128, 0, stream>>>(qkv_h, relf, sbuf_h);         // sbuf = owin f16
    { dim3 g(2048, 2);
      gemm_f16_k<0, 0><<<g, 256, 0, stream>>>(sbuf_h, wwlin16, b_wlin, nullptr, spa_w, 128, 128); }

    // 2) channel branch
    ln_f16_k<<<32768, 256, 0, stream>>>(x, ln2_g, ln2_b, xn_h);
    { dim3 g(2048, 2);
      gemm_f16_k<1, 0><<<g, 256, 0, stream>>>(xn_h, wq16, nullptr, nullptr, qi_h, 128, 128);
      gemm_f16_k<1, 0><<<g, 256, 0, stream>>>(xn_h, wk16, nullptr, nullptr, ki_h, 128, 128);
      gemm_f16_k<1, 0><<<g, 256, 0, stream>>>(xn_h, wv16, nullptr, nullptr, vi_h, 128, 128); }
    sumsq_k<<<512, 128, 0, stream>>>(qi_h, ki_h, qss, kss);
    gram_k<<<1024, 256, 0, stream>>>(qi_h, ki_h, G);
    attn_softmax_k<<<16, 32, 0, stream>>>(G, qss, kss, rescale, attn);
    apply_attn_k<<<131072, 256, 0, stream>>>(vi_h, attn, sbuf_h);      // sbuf = xo f16
    { dim3 g(2048, 2);
      gemm_f16_k<0, 0><<<g, 256, 0, stream>>>(sbuf_h, wproj16, b_proj, nullptr, specf, 128, 128); }
    dwconv_k<1><<<131072, 256, 0, stream>>>(vi_h, conv1_w, sbuf_h, nullptr);   // sbuf = t f16
    dwconv_k<0><<<131072, 256, 0, stream>>>(sbuf_h, conv2_w, nullptr, specf);  // spec += conv2

    // 3) residual + LN3 + MLP
    combine_ln3_k<<<32768, 256, 0, stream>>>(x, spa_w, specf, ln3_g, ln3_b, x2f, sbuf_h); // sbuf = h3 f16
    { dim3 g(2048, 8);
      gemm_f16_k<1, 1><<<g, 256, 0, stream>>>(sbuf_h, wmlp1_16, b_mlp1, nullptr, m1_h, 512, 128); }
    { dim3 g(2048, 2);
      gemm_f16_k<0, 0><<<g, 256, 0, stream>>>(m1_h, wmlp2_16, b_mlp2, x2f, (float*)d_out, 128, 512); }
}
